// SlidingRNN_31336081392293
// MI455X (gfx1250) — matrix operand order repaired and checked by reading, not compiled, not run on hardware
//
#include <hip/hip_runtime.h>
#include <hip/hip_bf16.h>

#define Bsz 512
#define Tsz 128
#define Csz 66
#define CszP 96
#define Hsz 512
#define Wsz 10
#define G3H 1536   // 3*H

typedef __attribute__((ext_vector_type(16))) __bf16 v16bf;
typedef __attribute__((ext_vector_type(8)))  float  v8f;

union Frag { v16bf v; uint4 q[2]; };

__device__ __forceinline__ unsigned short f2bf(float f) {
    unsigned int u = __float_as_uint(f);
    unsigned int r = (u + 0x7FFFu + ((u >> 16) & 1u)) >> 16;
    return (unsigned short)r;
}
__device__ __forceinline__ float bf2f(unsigned short s) {
    return __uint_as_float(((unsigned int)s) << 16);
}

// ---------------------------------------------------------------------------
// Pack fp32 weight matrix W (3H x K, row-major) into bf16 WMMA B-fragment
// layout: dst[ntile][ktile][lane][16], where B(k,n) = W(ntile*16 + n, ktile*32 + k)
// lane<16 covers k 0..7,16..23 of the chunk, lane>=16 covers k 8..15,24..31; n = lane&15.
// ---------------------------------------------------------------------------
__global__ void pack_weights_kernel(const float* __restrict__ Wsrc, int K, int KP,
                                    unsigned short* __restrict__ dst, int total) {
    int i = blockIdx.x * blockDim.x + threadIdx.x;
    if (i >= total) return;
    int idx   = i & 15;
    int lane  = (i >> 4) & 31;
    int flat  = i >> 9;              // ntile*(KP/32) + ktile
    int nkt   = KP >> 5;
    int kt    = flat % nkt;
    int ntile = flat / nkt;
    int p  = idx & 1;
    int vg = idx >> 1;
    int k = kt * 32 + ((lane >> 4) * 8) + ((idx >> 3) * 16) + (idx & 7);
    int n = ntile * 16 + (lane & 15);
    float v = (k < K) ? Wsrc[(size_t)n * K + k] : 0.0f;
    dst[i] = f2bf(v);
}

// pack x (B,T,C) fp32 -> (B,T,CszP) bf16, zero padded
__global__ void pack_x_kernel(const float* __restrict__ x, unsigned short* __restrict__ dst,
                              int total) {
    int i = blockIdx.x * blockDim.x + threadIdx.x;
    if (i >= total) return;
    int c  = i % CszP;
    int bt = i / CszP;
    dst[i] = (c < Csz) ? f2bf(x[(size_t)bt * Csz + c]) : (unsigned short)0;
}

__global__ void fill_zero_u16_kernel(unsigned short* __restrict__ p, int n) {
    int i = blockIdx.x * blockDim.x + threadIdx.x;
    if (i < n) p[i] = 0;
}

// ---------------------------------------------------------------------------
// One GRU cell step, all-bf16 data path.
// A sources are row-major bf16 (K-contiguous), so each lane's A fragment is two
// aligned 16B chunks -> 2x global_load_b128, no LDS staging.
// WG = 256 threads = 8 waves covering 64(B) x 128(H): wave wr in {0,1} picks a
// 32-row block, wc in {0..3} picks a 32-col block. Per wave: 2 M-tiles x 3 gates
// x 2 N-tiles = 12 f32 accumulators; each B fragment feeds 2 WMMAs.
// ---------------------------------------------------------------------------
__global__ __launch_bounds__(256) void gru_cell_kernel(
    const unsigned short* __restrict__ Abf_x, int ldx, int KxP,
    const unsigned short* __restrict__ WihF,
    const unsigned short* __restrict__ Hbf,        // B x H bf16 (A source + blend)
    const unsigned short* __restrict__ WhhF,
    const float* __restrict__ bih, const float* __restrict__ bhh,
    unsigned short* __restrict__ Hnew,             // B x H bf16
    float* __restrict__ ring) {                    // optional B x H fp32 window slot

    const int tid  = threadIdx.x;
    const int lane = tid & 31;
    const int wave = tid >> 5;
    const int wr   = wave >> 2;                    // 0..1 : 32-row block
    const int wc   = wave & 3;                     // 0..3 : 32-col block
    const int j0   = blockIdx.x * 128 + wc * 32;   // column base within H
    const int b0   = blockIdx.y * 64 + wr * 32;    // batch-row base for this wave

    v8f zero = {};
    v8f acc_i[2][3][2], acc_h[2][3][2];            // [mtile][gate][ntile]
    for (int m = 0; m < 2; ++m)
        for (int g = 0; g < 3; ++g)
            for (int n = 0; n < 2; ++n) { acc_i[m][g][n] = zero; acc_h[m][g][n] = zero; }

    const int khalf = (lane >> 4) * 8;             // lane<16 -> k 0..7/16..23 ; lane>=16 -> 8..15/24..31

    auto panel = [&](const unsigned short* __restrict__ Ab, int ldA, int KP,
                     const unsigned short* __restrict__ WF, v8f (&acc)[2][3][2]) {
        const int nkt = KP >> 5;
        const size_t ntStride = (size_t)nkt * 512;           // ushorts per ntile block
        const unsigned short* ar0 = Ab + (size_t)(b0 + (lane & 15)) * ldA + khalf;
        const unsigned short* ar1 = ar0 + (size_t)16 * ldA;
        for (int kt = 0; kt < nkt; ++kt) {
            Frag a0, a1;
            const unsigned short* p0 = ar0 + kt * 32;
            const unsigned short* p1 = ar1 + kt * 32;
            a0.q[0] = *(const uint4*)(p0);
            a0.q[1] = *(const uint4*)(p0 + 16);
            a1.q[0] = *(const uint4*)(p1);
            a1.q[1] = *(const uint4*)(p1 + 16);
            #pragma unroll
            for (int g = 0; g < 3; ++g) {
                #pragma unroll
                for (int nt = 0; nt < 2; ++nt) {
                    int ntile = (g * Hsz + j0 + nt * 16) >> 4;
                    const unsigned short* fp =
                        WF + (size_t)ntile * ntStride + (size_t)kt * 512 + lane * 16;
                    Frag bf;
                    bf.q[0] = *(const uint4*)(fp);
                    bf.q[1] = *(const uint4*)(fp + 8);
                    __builtin_prefetch(fp + 512, 0, 0);      // next K-tile of this fragment
                    acc[0][g][nt] = __builtin_amdgcn_wmma_f32_16x16x32_bf16(
                        false, a0.v, false, bf.v, (short)0, acc[0][g][nt], false, false);
                    acc[1][g][nt] = __builtin_amdgcn_wmma_f32_16x16x32_bf16(
                        false, a1.v, false, bf.v, (short)0, acc[1][g][nt], false, false);
                }
            }
        }
    };

    panel(Abf_x, ldx, KxP, WihF, acc_i);   // gi = x @ W_ih^T
    panel(Hbf,   Hsz, Hsz, WhhF, acc_h);   // gh = h @ W_hh^T

    // epilogue: fused GRU gates; C/D layout: vgpr e -> M = e (+8 for upper lanes)
    #pragma unroll
    for (int nt = 0; nt < 2; ++nt) {
        int col = j0 + nt * 16 + (lane & 15);
        float bi_r = bih[col],           bh_r = bhh[col];
        float bi_z = bih[Hsz + col],     bh_z = bhh[Hsz + col];
        float bi_n = bih[2 * Hsz + col], bh_n = bhh[2 * Hsz + col];
        #pragma unroll
        for (int m = 0; m < 2; ++m) {
            #pragma unroll
            for (int e = 0; e < 8; ++e) {
                int row = b0 + m * 16 + e + ((lane >> 4) ? 8 : 0);
                float ir = acc_i[m][0][nt][e] + bi_r, hr = acc_h[m][0][nt][e] + bh_r;
                float iz = acc_i[m][1][nt][e] + bi_z, hz = acc_h[m][1][nt][e] + bh_z;
                float in_ = acc_i[m][2][nt][e] + bi_n, hn = acc_h[m][2][nt][e] + bh_n;
                float r = 1.0f / (1.0f + __expf(-(ir + hr)));
                float z = 1.0f / (1.0f + __expf(-(iz + hz)));
                float n = tanhf(in_ + r * hn);
                float hp = bf2f(Hbf[(size_t)row * Hsz + col]);
                float hv = (1.0f - z) * n + z * hp;
                Hnew[(size_t)row * Hsz + col] = f2bf(hv);
                if (ring) ring[(size_t)row * Hsz + col] = hv;
            }
        }
    }
}

// ---------------------------------------------------------------------------
// Decoder frame: t(b,h) = sum_w tw[w]*win(b,w,h) + tb ; frame = t @ sw^T + sb + dec_in
// One block per batch row. Ring layout: [slot][B][H]; window order = base..base+9.
// Writes fp32 frame into d_out and zero-padded bf16 frame for the next GRU step.
// ---------------------------------------------------------------------------
__global__ __launch_bounds__(256) void frame_kernel(
    const float* __restrict__ ring, int base,
    const float* __restrict__ tw, const float* __restrict__ tb,
    const float* __restrict__ sw, const float* __restrict__ sb,
    const unsigned short* __restrict__ dec_in, int ld_dec,
    unsigned short* __restrict__ dec_next,         // B x CszP bf16
    float* __restrict__ outp) {                    // pre-offset by t*Csz
    __shared__ float tvec[Hsz];
    int b = blockIdx.x;
    for (int h = threadIdx.x; h < Hsz; h += 256) {
        float s = tb[0];
        #pragma unroll
        for (int i = 0; i < Wsz; ++i) {
            int slot = base + i; if (slot >= Wsz) slot -= Wsz;
            s += tw[i] * ring[((size_t)slot * Bsz + b) * Hsz + h];
        }
        tvec[h] = s;
    }
    __syncthreads();
    int c = threadIdx.x;
    if (c < CszP) {
        float s = 0.0f;
        if (c < Csz) {
            const float* swr = sw + (size_t)c * Hsz;
            s = sb[c] + bf2f(dec_in[(size_t)b * ld_dec + c]);
            #pragma unroll 4
            for (int h = 0; h < Hsz; ++h) s += tvec[h] * swr[h];
            outp[(size_t)b * Tsz * Csz + c] = s;
        }
        dec_next[(size_t)b * CszP + c] = (c < Csz) ? f2bf(s) : (unsigned short)0;
    }
}

// ---------------------------------------------------------------------------

extern "C" void kernel_launch(void* const* d_in, const int* in_sizes, int n_in,
                              void* d_out, int out_size, void* d_ws, size_t ws_size,
                              hipStream_t stream) {
    const float* x    = (const float*)d_in[0];
    const float* Wih0 = (const float*)d_in[1];
    const float* Whh0 = (const float*)d_in[2];
    const float* bih0 = (const float*)d_in[3];
    const float* bhh0 = (const float*)d_in[4];
    const float* Wih1 = (const float*)d_in[5];
    const float* Whh1 = (const float*)d_in[6];
    const float* bih1 = (const float*)d_in[7];
    const float* bhh1 = (const float*)d_in[8];
    const float* tw   = (const float*)d_in[9];
    const float* tb   = (const float*)d_in[10];
    const float* sw   = (const float*)d_in[11];
    const float* sb   = (const float*)d_in[12];
    float* out = (float*)d_out;

    char* ws = (char*)d_ws;
    size_t off = 0;
    auto alloc = [&](size_t bytes) -> void* {
        void* p = ws + off;
        off += (bytes + 255) & ~(size_t)255;
        return p;
    };

    unsigned short* WihF0 = (unsigned short*)alloc((size_t)G3H * CszP * 2);
    unsigned short* WhhF0 = (unsigned short*)alloc((size_t)G3H * Hsz * 2);
    unsigned short* WihF1 = (unsigned short*)alloc((size_t)G3H * Hsz * 2);
    unsigned short* WhhF1 = (unsigned short*)alloc((size_t)G3H * Hsz * 2);
    unsigned short* xbf   = (unsigned short*)alloc((size_t)Bsz * Tsz * CszP * 2);
    unsigned short* h0a   = (unsigned short*)alloc((size_t)Bsz * Hsz * 2);
    unsigned short* h0b   = (unsigned short*)alloc((size_t)Bsz * Hsz * 2);
    unsigned short* h1a   = (unsigned short*)alloc((size_t)Bsz * Hsz * 2);
    unsigned short* h1b   = (unsigned short*)alloc((size_t)Bsz * Hsz * 2);
    float*          ringB = (float*)alloc((size_t)Wsz * Bsz * Hsz * 4);
    unsigned short* fra   = (unsigned short*)alloc((size_t)Bsz * CszP * 2);
    unsigned short* frb   = (unsigned short*)alloc((size_t)Bsz * CszP * 2);

    // --- one-time prep -----------------------------------------------------
    {
        int t0 = G3H * CszP;
        pack_weights_kernel<<<(t0 + 255) / 256, 256, 0, stream>>>(Wih0, Csz, CszP, WihF0, t0);
        int t1 = G3H * Hsz;
        pack_weights_kernel<<<(t1 + 255) / 256, 256, 0, stream>>>(Whh0, Hsz, Hsz, WhhF0, t1);
        pack_weights_kernel<<<(t1 + 255) / 256, 256, 0, stream>>>(Wih1, Hsz, Hsz, WihF1, t1);
        pack_weights_kernel<<<(t1 + 255) / 256, 256, 0, stream>>>(Whh1, Hsz, Hsz, WhhF1, t1);
        int tx = Bsz * Tsz * CszP;
        pack_x_kernel<<<(tx + 255) / 256, 256, 0, stream>>>(x, xbf, tx);
        int nh = Bsz * Hsz;
        fill_zero_u16_kernel<<<(nh + 255) / 256, 256, 0, stream>>>(h0a, nh);
        fill_zero_u16_kernel<<<(nh + 255) / 256, 256, 0, stream>>>(h1a, nh);
    }

    dim3 gg(Hsz / 128, Bsz / 64);   // 4 x 8 workgroups, 256 threads each

    // --- encoder scan ------------------------------------------------------
    for (int t = 0; t < Tsz; ++t) {
        gru_cell_kernel<<<gg, 256, 0, stream>>>(
            xbf + (size_t)t * CszP, Tsz * CszP, CszP, WihF0,
            h0a, WhhF0, bih0, bhh0, h0b, nullptr);
        gru_cell_kernel<<<gg, 256, 0, stream>>>(
            h0b, Hsz, Hsz, WihF1,
            h1a, WhhF1, bih1, bhh1, h1b,
            ringB + (size_t)(t % Wsz) * Bsz * Hsz);
        unsigned short* tmp;
        tmp = h0a; h0a = h0b; h0b = tmp;
        tmp = h1a; h1a = h1b; h1b = tmp;
    }
    // window holds ys1[T-W..T-1]; oldest (t=118) sits in slot 118%10 == 8.

    // decoder step-0 input = x[:, T-1, :] -> alias into the packed-x buffer
    const unsigned short* dec_cur = xbf + (size_t)(Tsz - 1) * CszP;
    int ld_cur = Tsz * CszP;

    // --- decoder scan ------------------------------------------------------
    for (int d = 0; d < Tsz; ++d) {
        gru_cell_kernel<<<gg, 256, 0, stream>>>(
            dec_cur, ld_cur, CszP, WihF0,
            h0a, WhhF0, bih0, bhh0, h0b, nullptr);
        int newest = (8 + d) % Wsz;        // new h1 overwrites the oldest slot
        gru_cell_kernel<<<gg, 256, 0, stream>>>(
            h0b, Hsz, Hsz, WihF1,
            h1a, WhhF1, bih1, bhh1, h1b,
            ringB + (size_t)newest * Bsz * Hsz);
        int base = (newest + 1) % Wsz;     // oldest entry after the shift
        frame_kernel<<<Bsz, 256, 0, stream>>>(
            ringB, base, tw, tb, sw, sb, dec_cur, ld_cur, fra, out + (size_t)d * Csz);
        // next decoder input is the frame just produced
        dec_cur = fra; ld_cur = CszP;
        unsigned short* tmp;
        tmp = h0a; h0a = h0b; h0b = tmp;
        tmp = h1a; h1a = h1b; h1b = tmp;
        tmp = fra; fra = frb; frb = tmp;
    }
}
